// GrowthRotationF_3616362463458
// MI455X (gfx1250) — compile-verified, matrix operand order checked by reading, not run on hardware
//
#include <hip/hip_runtime.h>
#include <cstdint>

// ---------------- constants from the reference ----------------
#define BATCH 2048
#define NSTEP 128
#define HH    64
#define GAP   63
#define NR    64          // GAP+1
#define MCN   4096
#define DTF   (1.0f/128.0f)
#define PI_F  3.14159265358979323846f
#define SQDT  0.08838834764831845f   // sqrt(1/128)
#define SQ2D  1.0f                   // sqrt(2*DCOEF), DCOEF=0.5

typedef __attribute__((ext_vector_type(16))) _Float16 v16h;
typedef __attribute__((ext_vector_type(8)))  float    v8f;
typedef __attribute__((ext_vector_type(4)))  float    v4f;

// Native CDNA5 transcendental tanh if available (v_tanh_f32)
#if __has_builtin(__builtin_amdgcn_tanhf)
#define TANHF(x) __builtin_amdgcn_tanhf(x)
#elif __has_builtin(__builtin_amdgcn_tanh_f32)
#define TANHF(x) __builtin_amdgcn_tanh_f32(x)
#else
#define TANHF(x) tanhf(x)
#endif

__device__ __forceinline__ void wait_asynccnt0() {
#if __has_builtin(__builtin_amdgcn_s_wait_asynccnt)
  __builtin_amdgcn_s_wait_asynccnt(0);
#else
  asm volatile("s_wait_asynccnt 0" ::: "memory");
#endif
}

// Async DMA of `count` f32 from global to LDS (no format conversion needed).
// ldsOff is the byte offset of the destination inside the dynamic-LDS carve.
__device__ __forceinline__ void async_copy_f32(uint32_t ldsOff, const float* __restrict__ g, int count) {
  for (int i = threadIdx.x; i < count; i += 256) {
    uint32_t dst  = ldsOff + 4u * (uint32_t)i;
    uint32_t voff = 4u * (uint32_t)i;
    asm volatile("global_load_async_to_lds_b32 %0, %1, %2"
                 :: "v"(dst), "v"(voff), "s"((uint64_t)(uintptr_t)g)
                 : "memory");
  }
}

// ---------------- WMMA fragment swizzle helpers ----------------
// 16-bit A-fragment layout (ISA 7.12.2): lane L holds row (L&15), half g=L>>4;
// within a lane, element e of kk (k%32): g=(kk>>3)&1, e=((kk>>4)<<3)|(kk&7).
// A (M x 64) and B/W (64 x N) stored fragment-major so each lane reads one
// contiguous 32B v16h per ktile (2x ds_load_b128).
__device__ __forceinline__ int frag_idx(int tile, int kt, int laneRow, int kk) {
  int g = (kk >> 3) & 1;
  int e = ((kk >> 4) << 3) | (kk & 7);
  return (((tile * 2 + kt) * 32) + g * 16 + laneRow) * 16 + e;
}

__device__ __forceinline__ void storeAct(_Float16* S, int m, int k, float v) {
  S[frag_idx(m >> 4, k >> 5, m & 15, k & 31)] = (_Float16)v;
}

// Stage a [Krows x Ncols] f32 weight matrix (K padded to 64 with zeros) into
// LDS as f16 B-fragments; float4 global reads.
__device__ void stageW(const float* __restrict__ G, _Float16* S, int Krows, int Ncols) {
  int total4 = (64 * Ncols) >> 2;
  for (int i4 = threadIdx.x; i4 < total4; i4 += blockDim.x) {
    int i = i4 << 2;
    int k = i / Ncols;
    int n = i - k * Ncols;
    v4f v = {};
    if (k < Krows) v = *(const v4f*)(G + k * Ncols + n);
#pragma unroll
    for (int q = 0; q < 4; ++q)
      S[frag_idx((n + q) >> 4, k >> 5, (n + q) & 15, k & 31)] = (_Float16)v[q];
  }
}

// One 16x16 output tile, K=64 (2 x wmma 16x16x32 f16, f32 accum)
__device__ __forceinline__ v8f mmtile(const _Float16* A, const _Float16* W, int mt, int nt, int lane) {
  v8f acc = {};
#pragma unroll
  for (int kt = 0; kt < 2; ++kt) {
    v16h a = *(const v16h*)(A + ((mt * 2 + kt) * 32 + lane) * 16);
    v16h b = *(const v16h*)(W + ((nt * 2 + kt) * 32 + lane) * 16);
    acc = __builtin_amdgcn_wmma_f32_16x16x32_f16(false, a, false, b, (short)0, acc, false, false);
  }
  return acc;
}

// M=256, N=64 layer: Out = tanh(A @ W + bias), written back as A-fragments.
__device__ void gemm_tanh(const _Float16* A, const _Float16* W, const float* bias, _Float16* Out) {
  int wave = threadIdx.x >> 5, lane = threadIdx.x & 31;
  for (int t = wave; t < 64; t += 8) {      // 16 mtiles x 4 ntiles
    int mt = t >> 2, nt = t & 3;
    v8f acc = mmtile(A, W, mt, nt, lane);
    int n = nt * 16 + (lane & 15);
    float bv = bias[n];
    int mb = mt * 16 + ((lane >> 4) << 3);
#pragma unroll
    for (int i = 0; i < 8; ++i)
      storeAct(Out, mb + i, n, TANHF(acc[i] + bv));
  }
}

// ---------------- kernel 1: rate tables ----------------
__global__ __launch_bounds__(64) void rate_kernel(float* jm, float* cr, float* cfr) {
  int r = threadIdx.x;
  if (r >= NR) return;
  float rv = (float)(r + 2);
  float suml = 0.f, sumu = 0.f, c = 0.f;
  for (int k = 1; k <= GAP; ++k) {
    float l = (rv + (float)k <= 65.0f) ? 1.0f / (float)k : 0.0f;
    suml += l; c += l; jm[r * 126 + (k - 1)] = c;
  }
  for (int k = 1; k <= GAP; ++k) {
    float m = (rv - (float)k >= 2.0f) ? 1.0f / (float)k : 0.0f;
    sumu += m; c += m; jm[r * 126 + GAP + (k - 1)] = c;
  }
  float crv = c;
  for (int i = 0; i < 126; ++i) jm[r * 126 + i] /= crv;
  cr[r] = crv;
  cfr[r] = suml + sumu - crv;   // analytically ~0, kept for fidelity
}

// ---------------- kernel 2: MC jump expectation ----------------
__global__ __launch_bounds__(256) void mcjump_kernel(const float* __restrict__ rand_mc,
                                                     const float* __restrict__ jump_l,
                                                     const float* __restrict__ jm,
                                                     const float* __restrict__ cr,
                                                     float* __restrict__ mcj) {
  __shared__ float sjm[126];
  __shared__ int   hist[2][GAP];
  int r = blockIdx.x, t = threadIdx.x;
  if (t < 126) sjm[t] = jm[r * 126 + t];
  if (t < 2 * GAP) ((int*)hist)[t] = 0;
  __syncthreads();
  for (int mc = t; mc < MCN; mc += 256) {
    float rnd = rand_mc[mc * NR + r];
    int cnt = 0;
    for (int k = 0; k < 126; ++k) cnt += (rnd < sjm[k]) ? 1 : 0;
    int ind = 2 * GAP - cnt;
    int s = (ind < GAP) ? (ind + 1) : -(ind - GAP + 1);
    int a = (s < 0 ? -s : s) - 1;
    if (a > GAP - 1) a = GAP - 1;
    atomicAdd(&hist[(s < 0) ? 1 : 0][a], 1);
  }
  __syncthreads();
  float scale = cr[r] / (float)MCN;
  int c = t;  // 256 threads, 256 columns
  float acc = 0.f;
  for (int a = 0; a < GAP; ++a)
    acc += (float)hist[0][a] * jump_l[a * 256 + c] +
           (float)hist[1][a] * jump_l[(GAP + a) * 256 + c];
  mcj[r * 256 + c] = acc * scale;
}

// ---------------- kernel 3: initial u-net (runs once, scalar) ----------------
__global__ __launch_bounds__(64) void u0_kernel(const float* __restrict__ p, const int* __restrict__ rt0,
                                                const float* __restrict__ xt0, const float* __restrict__ emb_u,
                                                const float* __restrict__ Wu1, const float* __restrict__ bu1,
                                                const float* __restrict__ Wu2, const float* __restrict__ bu2,
                                                const float* __restrict__ Wu3, const float* __restrict__ bu3,
                                                const float* __restrict__ Wu4, const float* __restrict__ bu4,
                                                float* __restrict__ u0out) {
  __shared__ float h0[64][65];
  __shared__ float h1[64][65];
  int t = threadIdx.x;
  int b = blockIdx.x * 64 + t;
  float f0 = xt0[b * 3 + 0], f1 = xt0[b * 3 + 1], f2 = xt0[b * 3 + 2], f3 = p[b] * 0.025f;
  for (int j = 0; j < 64; ++j)
    h0[t][j] = TANHF(f0 * Wu1[j] + f1 * Wu1[64 + j] + f2 * Wu1[128 + j] + f3 * Wu1[192 + j] + bu1[j]);
  for (int j = 0; j < 64; ++j) {
    float a = bu2[j];
    for (int k = 0; k < 64; ++k) a += h0[t][k] * Wu2[k * 64 + j];
    h1[t][j] = TANHF(a);
  }
  for (int j = 0; j < 64; ++j) {
    float a = bu3[j];
    for (int k = 0; k < 64; ++k) a += h1[t][k] * Wu3[k * 64 + j];
    h0[t][j] = TANHF(a);
  }
  int ridx = rt0[b] - 2;
  float acc0 = 0.f, acc1 = 0.f;
  for (int c = 0; c < 256; ++c) {
    float a = bu4[c];
    for (int k = 0; k < 64; ++k) a += h0[t][k] * Wu4[k * 256 + c];
    float v = emb_u[ridx * 256 + c] * a;
    if (c < 128) acc0 += v; else acc1 += v;
  }
  u0out[b * 2 + 0] = acc0;
  u0out[b * 2 + 1] = acc1;
}

// ---------------- kernel 4: persistent time-step simulation ----------------
// Dynamic-LDS carve (byte offsets; all 32B aligned)
#define OFF_AX    0u
#define OFF_AY    32768u
#define OFF_W1    65536u
#define OFF_W2    73728u
#define OFF_W3    81920u
#define OFF_W4G   90112u
#define OFF_W4J   188416u
#define OFF_W1J   221184u
#define OFF_W2J   229376u
#define OFF_W3J   237568u
#define OFF_B1    245760u
#define OFF_B2    246016u
#define OFF_B3    246272u
#define OFF_B4G   246528u
#define OFF_B1J   249600u
#define OFF_B2J   249856u
#define OFF_B3J   250112u
#define OFF_B4J   250368u
#define OFF_SGU   251392u
#define OFF_SJ    257536u
#define OFF_SRIDX 259584u
#define OFF_SDRT  260608u
#define SMEM_BYTES 261632u

struct SimP {
  const float *p, *xt0, *dBt, *rjump, *rjsize;
  const int*   rt0;
  const float *emb_g;
  const float *Wg1, *bg1, *Wg2, *bg2, *Wg3, *bg3, *Wg4, *bg4;
  const float *Wj1, *bj1, *Wj2, *bj2, *Wj3, *bj3, *Wj4, *bj4;
  const float *jump_r, *jump_l;
  const float *jm, *cr, *cfr, *mcj, *u0;
  float* out;
};

__global__ __launch_bounds__(256) void sim_kernel(SimP P) {
  extern __shared__ char smem[];   // keep addrspace(3): constant-offset typed views only
  _Float16* aX  = (_Float16*)(smem + OFF_AX);
  _Float16* aY  = (_Float16*)(smem + OFF_AY);
  _Float16* w1  = (_Float16*)(smem + OFF_W1);
  _Float16* w2  = (_Float16*)(smem + OFF_W2);
  _Float16* w3  = (_Float16*)(smem + OFF_W3);
  _Float16* w4g = (_Float16*)(smem + OFF_W4G);
  _Float16* w4j = (_Float16*)(smem + OFF_W4J);
  _Float16* w1j = (_Float16*)(smem + OFF_W1J);
  _Float16* w2j = (_Float16*)(smem + OFF_W2J);
  _Float16* w3j = (_Float16*)(smem + OFF_W3J);
  float* b1  = (float*)(smem + OFF_B1);
  float* b2  = (float*)(smem + OFF_B2);
  float* b3  = (float*)(smem + OFF_B3);
  float* b4g = (float*)(smem + OFF_B4G);
  float* b1j = (float*)(smem + OFF_B1J);
  float* b2j = (float*)(smem + OFF_B2J);
  float* b3j = (float*)(smem + OFF_B3J);
  float* b4j = (float*)(smem + OFF_B4J);
  float* sGU = (float*)(smem + OFF_SGU);
  float* sJ  = (float*)(smem + OFF_SJ);
  int* sRidx = (int*)(smem + OFF_SRIDX);
  int* sDrt  = (int*)(smem + OFF_SDRT);

  const int t = threadIdx.x;
  const int b = blockIdx.x * 256 + t;
  const int wave = t >> 5, lane = t & 31;

  // per-row register state
  float pv  = P.p[b];
  float p40 = pv * 0.025f;
  float xt[3]  = {P.xt0[b * 3], P.xt0[b * 3 + 1], P.xt0[b * 3 + 2]};
  float xti[3] = {xt[0], xt[1], xt[2]};
  float gt[3][3] = {{1.f,0.f,0.f},{0.f,1.f,0.f},{0.f,0.f,1.f}};
  int   rt = P.rt0[b];
  float funr = 0.f, funi = 0.f;
  float upr = P.u0[b * 2], upi = P.u0[b * 2 + 1];

  for (int step = 0; step < NSTEP; ++step) {
    // async DMA: this step's bias vectors straight into LDS (ASYNCcnt path)
    async_copy_f32(OFF_B1,  P.bg1 + step * 64,  64);
    async_copy_f32(OFF_B2,  P.bg2 + step * 64,  64);
    async_copy_f32(OFF_B3,  P.bg3 + step * 64,  64);
    async_copy_f32(OFF_B4G, P.bg4 + step * 768, 768);
    async_copy_f32(OFF_B1J, P.bj1 + step * 64,  64);
    async_copy_f32(OFF_B2J, P.bj2 + step * 64,  64);
    async_copy_f32(OFF_B3J, P.bj3 + step * 64,  64);
    async_copy_f32(OFF_B4J, P.bj4 + step * 256, 256);

    // prefetch next step's large weight matrices into cache
    if (step + 1 < NSTEP) {
      const char* ng = (const char*)(P.Wg4 + (size_t)(step + 1) * (HH * 768));
      for (int i = t; i < 1536; i += 256) __builtin_prefetch(ng + i * 128, 0, 1);
      const char* nj = (const char*)(P.Wj4 + (size_t)(step + 1) * (HH * 256));
      for (int i = t; i < 512; i += 256) __builtin_prefetch(nj + i * 128, 0, 1);
    }
    // stage this step's weights f32 -> f16 fragments in LDS
    stageW(P.Wg1 + (size_t)step * 4 * 64,   w1,  4, 64);
    stageW(P.Wg2 + (size_t)step * 64 * 64,  w2, 64, 64);
    stageW(P.Wg3 + (size_t)step * 64 * 64,  w3, 64, 64);
    stageW(P.Wg4 + (size_t)step * 64 * 768, w4g, 64, 768);
    stageW(P.Wj1 + (size_t)step * 4 * 64,   w1j, 4, 64);
    stageW(P.Wj2 + (size_t)step * 64 * 64,  w2j, 64, 64);
    stageW(P.Wj3 + (size_t)step * 64 * 64,  w3j, 64, 64);
    stageW(P.Wj4 + (size_t)step * 64 * 256, w4j, 64, 256);

    // phase0: jump decision + feature staging (one row per thread)
    int ridx = rt - 2;
    float crv = P.cr[ridx];
    float rj  = P.rjump[step * BATCH + b];
    float rjs = P.rjsize[step * BATCH + b];
    const float* jrow = P.jm + ridx * 126;
    int cnt = 0;
    for (int k = 0; k < 126; ++k) cnt += (rjs < jrow[k]) ? 1 : 0;
    int ind = 2 * GAP - cnt;
    int sz  = (ind < GAP) ? (ind + 1) : -(ind - GAP + 1);
    int drt = (rj < crv * DTF) ? sz : 0;
    sRidx[t] = ridx; sDrt[t] = drt;
#pragma unroll
    for (int j = 0; j < 6; ++j) sGU[t * 6 + j] = 0.f;
    sJ[t * 2] = 0.f; sJ[t * 2 + 1] = 0.f;
    storeAct(aX, t, 0, xti[0]); storeAct(aX, t, 1, xti[1]);
    storeAct(aX, t, 2, xti[2]); storeAct(aX, t, 3, p40);
    for (int k = 4; k < 64; ++k) storeAct(aX, t, k, 0.f);
    wait_asynccnt0();           // biases landed in LDS
    __syncthreads();

    // ---- g-network ----
    gemm_tanh(aX, w1, b1, aY);  __syncthreads();
    gemm_tanh(aY, w2, b2, aX);  __syncthreads();
    gemm_tanh(aX, w3, b3, aY);  __syncthreads();
    {
      const float* embS = P.emb_g + (size_t)step * NR * 768;
      for (int tt = wave; tt < 16 * 48; tt += 8) {
        int mt = tt / 48, nt = tt - mt * 48;
        v8f acc = mmtile(aY, w4g, mt, nt, lane);
        int n = nt * 16 + (lane & 15);
        float bv = b4g[n];
        int comp = n >> 7;
        int mb = mt * 16 + ((lane >> 4) << 3);
#pragma unroll
        for (int i = 0; i < 8; ++i) {
          int m = mb + i;
          float w = embS[(size_t)sRidx[m] * 768 + n];
          atomicAdd(&sGU[m * 6 + comp], (acc[i] + bv) * w);
        }
      }
    }
    __syncthreads();

    // ---- j-network (re-stage features) ----
    storeAct(aX, t, 0, xti[0]); storeAct(aX, t, 1, xti[1]);
    storeAct(aX, t, 2, xti[2]); storeAct(aX, t, 3, p40);
    for (int k = 4; k < 64; ++k) storeAct(aX, t, k, 0.f);
    __syncthreads();
    gemm_tanh(aX, w1j, b1j, aY);  __syncthreads();
    gemm_tanh(aY, w2j, b2j, aX);  __syncthreads();
    gemm_tanh(aX, w3j, b3j, aY);  __syncthreads();
    {
      for (int tt = wave; tt < 16 * 16; tt += 8) {
        int mt = tt >> 4, nt = tt & 15;
        v8f acc = mmtile(aY, w4j, mt, nt, lane);
        int n = nt * 16 + (lane & 15);
        float bv = b4j[n];
        int comp = n >> 7;
        int mb = mt * 16 + ((lane >> 4) << 3);
#pragma unroll
        for (int i = 0; i < 8; ++i) {
          int m = mb + i;
          int rr = sRidx[m], dd = sDrt[m];
          int ad = dd < 0 ? -dd : dd;
          float jl = 0.f;
          if (ad > 0) {
            int ai = ad - 1; if (ai > GAP - 1) ai = GAP - 1;
            jl = (dd > 0) ? P.jump_l[ai * 256 + n] : P.jump_l[(GAP + ai) * 256 + n];
          }
          float w = P.jump_r[rr * 256 + n] * (jl - P.mcj[rr * 256 + n] * DTF);
          atomicAdd(&sJ[m * 2 + comp], (acc[i] + bv) * w);
        }
      }
    }
    __syncthreads();

    // phase1: per-row geometry + accumulator update
    {
      float gu[2][3];
#pragma unroll
      for (int j = 0; j < 6; ++j) gu[j / 3][j % 3] = sGU[t * 6 + j];
      float jcr = sJ[t * 2], jci = sJ[t * 2 + 1];
      float rnorm = sqrtf(xt[0]*xt[0] + xt[1]*xt[1] + xt[2]*xt[2]);
      float cz = fminf(1.f, fmaxf(-1.f, xt[2] / rnorm));
      float theta = acosf(cz);
      float phi = atan2f(xt[1], xt[0]);
      float th = theta - 0.5f * PI_F;
      float ct = cosf(th), st = sinf(th), cp = cosf(phi), sp = sinf(phi);
      float Ti[3][3] = {{cp*ct, -sp, cp*st},{sp*ct, cp, sp*st},{-st, 0.f, ct}};
      float M[3][3];
#pragma unroll
      for (int i2 = 0; i2 < 3; ++i2)
#pragma unroll
        for (int k2 = 0; k2 < 3; ++k2)
          M[i2][k2] = gt[i2][0]*Ti[0][k2] + gt[i2][1]*Ti[1][k2] + gt[i2][2]*Ti[2][k2];
      float g2[2][3];
#pragma unroll
      for (int i2 = 0; i2 < 2; ++i2)
#pragma unroll
        for (int k2 = 0; k2 < 3; ++k2)
          g2[i2][k2] = gu[i2][0]*M[0][k2] + gu[i2][1]*M[1][k2] + gu[i2][2]*M[2][k2];
      float db0 = P.dBt[((size_t)step * BATCH + b) * 2 + 0] * SQDT;
      float db1 = P.dBt[((size_t)step * BATCH + b) * 2 + 1] * SQDT;
      float rtf = (float)rt;
      float sc  = SQ2D / rtf;
      // guc_e0 = -guc[2], guc_e1 = guc[1] (gep contraction)
      float dotr = -g2[0][2]*db0 + g2[0][1]*db1;
      float doti = -g2[1][2]*db0 + g2[1][1]*db1;
      float incr_r = sc * dotr + jcr;
      float incr_i = sc * doti + jci;
      float ea  = expf(-funr);
      float efr = ea * cosf(funi), efi = -ea * sinf(funi);   // exp(-fun)
      upr += efr * incr_r - efi * incr_i;
      upi += efr * incr_i + efi * incr_r;
      float Uv = 1.f / (1.f + rtf * rnorm);                  // uses pre-update xt
      funr += P.cfr[ridx] * DTF;
      funi += pv * Uv * DTF;
      float dth = sc * db0 + 0.5f * PI_F;
      float dph = sc * db1;
      float sd = sinf(dth), cd = cosf(dth), cph2 = cosf(dph), sph2 = sinf(dph);
      float tx0 = sd * cph2 - 1.f, tx1 = sd * sph2, tx2 = cd;
      float dX3[3];
#pragma unroll
      for (int i2 = 0; i2 < 3; ++i2)
        dX3[i2] = Ti[i2][0]*tx0 + Ti[i2][1]*tx1 + Ti[i2][2]*tx2;
#pragma unroll
      for (int i2 = 0; i2 < 3; ++i2) xt[i2] += dX3[i2];
#pragma unroll
      for (int i2 = 0; i2 < 3; ++i2)
        xti[i2] += gt[i2][0]*dX3[0] + gt[i2][1]*dX3[1] + gt[i2][2]*dX3[2];
      rt += drt;
      float nrm = sqrtf(xti[0]*xti[0] + xti[1]*xti[1] + xti[2]*xti[2]);
      if (nrm > 1.f) {
        float nb[3] = {xti[0]/nrm, xti[1]/nrm, xti[2]/nrm};
        float d = -((nb[0]-xti[0])*nb[0] + (nb[1]-xti[1])*nb[1] + (nb[2]-xti[2])*nb[2]);
#pragma unroll
        for (int i2 = 0; i2 < 3; ++i2) xti[i2] -= 2.f * d * nb[i2];
        float tmp[3];
#pragma unroll
        for (int k2 = 0; k2 < 3; ++k2)
          tmp[k2] = nb[0]*gt[0][k2] + nb[1]*gt[1][k2] + nb[2]*gt[2][k2];
#pragma unroll
        for (int i2 = 0; i2 < 3; ++i2)
#pragma unroll
          for (int k2 = 0; k2 < 3; ++k2)
            gt[i2][k2] -= 2.f * nb[i2] * tmp[k2];
      }
    }
    __syncthreads();
  }

  // final outputs
  float nrm = sqrtf(xti[0]*xti[0] + xti[1]*xti[1] + xti[2]*xti[2]);
  float Uv  = 1.f / (1.f + (float)rt * nrm);
  float phc = pv * Uv - funi;
  float ea  = expf(-funr);
  P.out[b * 2 + 0] = upr;
  P.out[b * 2 + 1] = upi;
  P.out[2 * BATCH + b * 2 + 0] = ea * cosf(phc);
  P.out[2 * BATCH + b * 2 + 1] = ea * sinf(phc);
}

// ---------------- host launcher ----------------
extern "C" void kernel_launch(void* const* d_in, const int* in_sizes, int n_in,
                              void* d_out, int out_size, void* d_ws, size_t ws_size,
                              hipStream_t stream) {
  (void)in_sizes; (void)n_in; (void)out_size; (void)ws_size;
  const float* p      = (const float*)d_in[0];
  const int*   rt0    = (const int*)  d_in[1];
  const float* xt0    = (const float*)d_in[2];
  const float* dBt    = (const float*)d_in[3];
  const float* rjump  = (const float*)d_in[4];
  const float* rjsize = (const float*)d_in[5];
  const float* rand_mc= (const float*)d_in[6];
  const float* emb_u  = (const float*)d_in[7];
  const float* Wu1 = (const float*)d_in[8];  const float* bu1 = (const float*)d_in[9];
  const float* Wu2 = (const float*)d_in[10]; const float* bu2 = (const float*)d_in[11];
  const float* Wu3 = (const float*)d_in[12]; const float* bu3 = (const float*)d_in[13];
  const float* Wu4 = (const float*)d_in[14]; const float* bu4 = (const float*)d_in[15];
  const float* emb_g  = (const float*)d_in[16];
  const float* Wg1 = (const float*)d_in[17]; const float* bg1 = (const float*)d_in[18];
  const float* Wg2 = (const float*)d_in[19]; const float* bg2 = (const float*)d_in[20];
  const float* Wg3 = (const float*)d_in[21]; const float* bg3 = (const float*)d_in[22];
  const float* Wg4 = (const float*)d_in[23]; const float* bg4 = (const float*)d_in[24];
  const float* Wj1 = (const float*)d_in[25]; const float* bj1 = (const float*)d_in[26];
  const float* Wj2 = (const float*)d_in[27]; const float* bj2 = (const float*)d_in[28];
  const float* Wj3 = (const float*)d_in[29]; const float* bj3 = (const float*)d_in[30];
  const float* Wj4 = (const float*)d_in[31]; const float* bj4 = (const float*)d_in[32];
  const float* jump_r = (const float*)d_in[33];
  const float* jump_l = (const float*)d_in[34];

  // workspace carve
  float* ws  = (float*)d_ws;
  float* jm  = ws;              // 64*126
  float* cr  = jm  + 64 * 126;  // 64
  float* cfr = cr  + 64;        // 64
  float* mcj = cfr + 64;        // 64*256
  float* u0  = mcj + 64 * 256;  // 2048*2

  rate_kernel<<<1, 64, 0, stream>>>(jm, cr, cfr);
  mcjump_kernel<<<64, 256, 0, stream>>>(rand_mc, jump_l, jm, cr, mcj);
  u0_kernel<<<BATCH / 64, 64, 0, stream>>>(p, rt0, xt0, emb_u, Wu1, bu1, Wu2, bu2,
                                           Wu3, bu3, Wu4, bu4, u0);
  SimP P;
  P.p = p; P.xt0 = xt0; P.dBt = dBt; P.rjump = rjump; P.rjsize = rjsize; P.rt0 = rt0;
  P.emb_g = emb_g;
  P.Wg1 = Wg1; P.bg1 = bg1; P.Wg2 = Wg2; P.bg2 = bg2; P.Wg3 = Wg3; P.bg3 = bg3;
  P.Wg4 = Wg4; P.bg4 = bg4;
  P.Wj1 = Wj1; P.bj1 = bj1; P.Wj2 = Wj2; P.bj2 = bj2; P.Wj3 = Wj3; P.bj3 = bj3;
  P.Wj4 = Wj4; P.bj4 = bj4;
  P.jump_r = jump_r; P.jump_l = jump_l;
  P.jm = jm; P.cr = cr; P.cfr = cfr; P.mcj = mcj; P.u0 = u0;
  P.out = (float*)d_out;

  sim_kernel<<<BATCH / 256, 256, SMEM_BYTES, stream>>>(P);
}
